// PointCompressor_23570780520797
// MI455X (gfx1250) — compile-verified, matrix operand order checked by reading, not run on hardware
//
#include <hip/hip_runtime.h>
#include <hip/hip_bf16.h>

// ---------------------------------------------------------------------------
// PointCompressor forward for MI455X (gfx1250, wave32).
// WMMA f16->f32 GEMMs (64x128 block tile, fragment-major LDS staging),
// async global->LDS tiling for the O(N^2) KNN / Chamfer scans,
// scalar loss assembled via block reductions.
// ---------------------------------------------------------------------------

typedef __attribute__((ext_vector_type(16))) _Float16 v16h;
typedef __attribute__((ext_vector_type(8)))  float    v8f;
typedef __attribute__((ext_vector_type(4)))  unsigned uint4v;

// ------------------------- CDNA5 async global->LDS -------------------------
__device__ __forceinline__ void async_g2l_b32(float* lds_dst, const float* gsrc) {
  // Low 32 bits of a generic LDS pointer are the LDS byte address (ISA 10.2).
  asm volatile("global_load_async_to_lds_b32 %0, %1, off"
               :: "v"((unsigned)(unsigned long long)(size_t)lds_dst),
                  "v"((unsigned long long)(size_t)gsrc)
               : "memory");
}
__device__ __forceinline__ void wait_async0() {
  asm volatile("s_wait_asynccnt 0" ::: "memory");
}

// --------------------------- fragment helpers ------------------------------
__device__ __forceinline__ unsigned pack2h(float a, float b) {
  auto p = __builtin_amdgcn_cvt_pkrtz(a, b);      // v_cvt_pk_rtz_f16_f32 (__fp16 x2)
  return __builtin_bit_cast(unsigned, p);
}
__device__ __forceinline__ v16h load_frag(const unsigned* p) {
  union { struct { uint4v a, b; } s; v16h h; } u; // 2x ds_load_b128
  u.s.a = *(const uint4v*)(p);
  u.s.b = *(const uint4v*)(p + 4);
  return u.h;
}

// ------------------------------ WMMA GEMM ----------------------------------
// C[M,N] = act(A[M,K] * W[K,N] + bias + residual). A,W row-major f32.
// Block: 256 threads = 8 waves; 64(M) x 128(N) tile, K chunks of 32.
// LDS holds tiles in the ISA 7.12.2 fragment-major layout:
//   per lane, 8 consecutive dwords == the lane's 16 packed f16 for one frag.
// Each wave: 1 B fragment + 4 A fragments -> 4 v_wmma per K chunk.
template<int ACT>
__global__ __launch_bounds__(256)
void pc_gemm_wmma(const float* __restrict__ A, const float* __restrict__ W,
                  const float* __restrict__ bias, const float* __restrict__ Res,
                  float* __restrict__ C, int M, int Kd, int Nc)
{
  __shared__ __align__(16) unsigned As_frag[4][32][8];
  __shared__ __align__(16) unsigned Ws_frag[8][32][8];
  const int tid  = threadIdx.x;
  const int wv_  = tid >> 5;
  const int lane = tid & 31;
  const int hh   = lane >> 4;
  const int nn   = lane & 15;
  const int row0 = blockIdx.x * 64;
  const int col0 = blockIdx.y * 128;
  v8f acc[4] = {};
  for (int k0 = 0; k0 < Kd; k0 += 32) {
    const bool fullk = (k0 + 32 <= Kd);
    const bool aFull = fullk && (row0 + 64 <= M);
    const bool wFull = fullk && (col0 + 128 <= Nc);
    // ---- stage A: 64 rows x 16 k-pairs = 1024 words, 4 per thread
    #pragma unroll
    for (int u = 0; u < 4; ++u) {
      int i = tid + u * 256;
      int r = i >> 4, t = i & 15;            // row in tile, k-pair
      int gr = row0 + r, gk = k0 + 2 * t;
      float x0, x1;
      if (aFull) {
        const float* p = A + (size_t)gr * Kd + gk;
        x0 = p[0]; x1 = p[1];
      } else {
        x0 = (gr < M && gk     < Kd) ? A[(size_t)gr * Kd + gk]     : 0.f;
        x1 = (gr < M && gk + 1 < Kd) ? A[(size_t)gr * Kd + gk + 1] : 0.f;
      }
      int h = (t >> 2) & 1;
      int v = ((t >> 3) << 2) | (t & 3);
      As_frag[r >> 4][(r & 15) + 16 * h][v] = pack2h(x0, x1);
    }
    // ---- stage W: 16 k-pairs x 128 cols = 2048 words, 8 per thread
    #pragma unroll
    for (int u = 0; u < 8; ++u) {
      int i = tid + u * 256;
      int t = i >> 7, c = i & 127;           // k-pair, col in tile
      int gk = k0 + 2 * t, gc = col0 + c;
      float x0, x1;
      if (wFull) {
        x0 = W[(size_t)gk * Nc + gc];
        x1 = W[(size_t)(gk + 1) * Nc + gc];
      } else {
        x0 = (gk     < Kd && gc < Nc) ? W[(size_t)gk * Nc + gc]       : 0.f;
        x1 = (gk + 1 < Kd && gc < Nc) ? W[(size_t)(gk + 1) * Nc + gc] : 0.f;
      }
      int h = (t >> 2) & 1;
      int v = ((t >> 3) << 2) | (t & 3);
      Ws_frag[c >> 4][(c & 15) + 16 * h][v] = pack2h(x0, x1);
    }
    if (fullk && k0 + 32 < Kd) {             // next-chunk prefetch (speculative)
      __builtin_prefetch(A + (size_t)(row0 + (tid & 63)) * Kd + k0 + 32, 0, 1);
      __builtin_prefetch(W + (size_t)(k0 + 32 + (tid >> 7)) * Nc + col0 + (tid & 127), 0, 1);
    }
    __syncthreads();
    v16h bf = load_frag(&Ws_frag[wv_][lane][0]);
    #pragma unroll
    for (int mt = 0; mt < 4; ++mt) {
      v16h af = load_frag(&As_frag[mt][lane][0]);
      acc[mt] = __builtin_amdgcn_wmma_f32_16x16x32_f16(false, af, false, bf,
                                                       (short)0, acc[mt], false, false);
    }
    __syncthreads();
  }
  #pragma unroll
  for (int mt = 0; mt < 4; ++mt) {
    #pragma unroll
    for (int v = 0; v < 8; ++v) {
      int gr = row0 + mt * 16 + v + 8 * hh;
      int gc = col0 + wv_ * 16 + nn;
      if (gr < M && gc < Nc) {
        float val = acc[mt][v];
        if (bias) val += bias[gc];
        if (Res)  val += Res[(size_t)gr * Nc + gc];
        if (ACT == 1) val = fmaxf(val, 0.f);
        C[(size_t)gr * Nc + gc] = val;
      }
    }
  }
}

// ------------------------------- KNN top-16 --------------------------------
__global__ __launch_bounds__(256)
void pc_knn_topk(const float* __restrict__ qxyz, const float* __restrict__ rxyz,
                 int nq, int nref, int* __restrict__ out)
{
  __shared__ float sref[256 * 3];
  const int b  = blockIdx.y;
  const int qi = blockIdx.x * 256 + threadIdx.x;
  const bool valid = qi < nq;
  float qx = 0.f, qy = 0.f, qz = 0.f;
  if (valid) {
    const float* qp = qxyz + ((size_t)b * nq + qi) * 3;
    qx = qp[0]; qy = qp[1]; qz = qp[2];
  }
  float bd[16]; int bi[16];
  #pragma unroll
  for (int t = 0; t < 16; ++t) { bd[t] = 3.0e38f; bi[t] = 0; }
  const float* rbase = rxyz + (size_t)b * nref * 3;
  for (int t0 = 0; t0 < nref; t0 += 256) {
    int tc  = nref - t0; if (tc > 256) tc = 256;
    int nel = tc * 3;
    for (int i = threadIdx.x; i < nel; i += 256)
      async_g2l_b32(&sref[i], rbase + (size_t)t0 * 3 + i);
    wait_async0();
    __syncthreads();
    if (valid) {
      for (int j = 0; j < tc; ++j) {
        float dx = qx - sref[j*3+0];
        float dy = qy - sref[j*3+1];
        float dz = qz - sref[j*3+2];
        float d  = dx*dx + dy*dy + dz*dz;
        if (d < bd[15]) {                    // register-resident insertion
          bd[15] = d; bi[15] = t0 + j;
          #pragma unroll
          for (int t = 15; t > 0; --t) {
            if (bd[t] < bd[t-1]) {
              float td = bd[t]; bd[t] = bd[t-1]; bd[t-1] = td;
              int   ti = bi[t]; bi[t] = bi[t-1]; bi[t-1] = ti;
            }
          }
        }
      }
    }
    __syncthreads();
  }
  if (valid) {
    int* op = out + ((size_t)b * nq + qi) * 16;
    #pragma unroll
    for (int t = 0; t < 16; ++t) op[t] = bi[t];
  }
}

// --------------------------- Chamfer (one side) ----------------------------
__global__ __launch_bounds__(256)
void pc_chamfer_min(const float* __restrict__ a, const float* __restrict__ ref,
                    int na, int nref, float* __restrict__ acc)
{
  __shared__ float sref[256 * 3];
  __shared__ float red[256];
  const int b  = blockIdx.y;
  const int qi = blockIdx.x * 256 + threadIdx.x;
  const bool valid = qi < na;
  float qx = 0.f, qy = 0.f, qz = 0.f;
  if (valid) {
    const float* qp = a + ((size_t)b * na + qi) * 3;
    qx = qp[0]; qy = qp[1]; qz = qp[2];
  }
  float mind = 3.0e38f;
  const float* rbase = ref + (size_t)b * nref * 3;
  for (int t0 = 0; t0 < nref; t0 += 256) {
    int tc  = nref - t0; if (tc > 256) tc = 256;
    int nel = tc * 3;
    for (int i = threadIdx.x; i < nel; i += 256)
      async_g2l_b32(&sref[i], rbase + (size_t)t0 * 3 + i);
    wait_async0();
    __syncthreads();
    if (valid) {
      for (int j = 0; j < tc; ++j) {
        float dx = qx - sref[j*3+0];
        float dy = qy - sref[j*3+1];
        float dz = qz - sref[j*3+2];
        mind = fminf(mind, dx*dx + dy*dy + dz*dz);
      }
    }
    __syncthreads();
  }
  red[threadIdx.x] = valid ? mind : 0.f;
  __syncthreads();
  for (int s2 = 128; s2 > 0; s2 >>= 1) {
    if (threadIdx.x < s2) red[threadIdx.x] += red[threadIdx.x + s2];
    __syncthreads();
  }
  if (threadIdx.x == 0) atomicAdd(acc, red[0]);
}

// ------------------------- small elementwise kernels -----------------------
__global__ void pc_zero(float* p, int n) {
  int i = blockIdx.x * 256 + threadIdx.x;
  if (i < n) p[i] = 0.f;
}
__global__ void pc_gather(const float* __restrict__ src, const int* __restrict__ idx,
                          float* __restrict__ dst, long R, int C, int Mq)
{
  long i = (long)blockIdx.x * 256 + threadIdx.x;
  if (i >= R * (long)C) return;
  long r = i / C; int c = (int)(i % C);
  long bm = r >> 4;
  int b = (int)(bm / Mq);
  int id = idx[r];
  dst[i] = src[((size_t)b * Mq + id) * C + c];
}
__global__ void pc_rel(const float* __restrict__ xyz, const int* __restrict__ idx,
                       float* __restrict__ rel, long R, int Mq)
{
  long i = (long)blockIdx.x * 256 + threadIdx.x;
  if (i >= R * 3) return;
  long r = i / 3; int c = (int)(i % 3);
  long bm = r >> 4;
  int b = (int)(bm / Mq), m2 = (int)(bm % Mq);
  int id = idx[r];
  rel[i] = xyz[((size_t)b * Mq + m2) * 3 + c] - xyz[((size_t)b * Mq + id) * 3 + c];
}
__global__ void pc_add(float* __restrict__ a, const float* __restrict__ b2, long n) {
  long i = (long)blockIdx.x * 256 + threadIdx.x;
  if (i < n) a[i] += b2[i];
}
__global__ void pc_qkp(float* __restrict__ kg, const float* __restrict__ q,
                       const float* __restrict__ pos, long R, int C)
{
  long i = (long)blockIdx.x * 256 + threadIdx.x;
  if (i >= R * (long)C) return;
  long bm = (i / C) >> 4;
  int c = (int)(i % C);
  kg[i] = q[(size_t)bm * C + c] - kg[i] + pos[i];
}
__global__ void pc_attend(const float* __restrict__ a, const float* __restrict__ vp,
                          float* __restrict__ out, long BM, int C)
{
  long i = (long)blockIdx.x * 256 + threadIdx.x;
  if (i >= BM * (long)C) return;
  long bm = i / C; int c = (int)(i % C);
  const float* ap  = a  + ((size_t)bm * 16) * C + c;
  const float* vpp = vp + ((size_t)bm * 16) * C + c;
  float mx = -3.0e38f;
  #pragma unroll
  for (int kk = 0; kk < 16; ++kk) mx = fmaxf(mx, ap[(size_t)kk * C]);
  float e[16]; float s = 0.f;
  #pragma unroll
  for (int kk = 0; kk < 16; ++kk) { e[kk] = __expf(ap[(size_t)kk * C] - mx); s += e[kk]; }
  float inv = 1.f / s, o = 0.f;
  #pragma unroll
  for (int kk = 0; kk < 16; ++kk) o += e[kk] * inv * vpp[(size_t)kk * C];
  out[i] = o;
}
__global__ void pc_maxk(const float* __restrict__ h, float* __restrict__ out, long BM, int C)
{
  long i = (long)blockIdx.x * 256 + threadIdx.x;
  if (i >= BM * (long)C) return;
  long bm = i / C; int c = (int)(i % C);
  const float* p = h + ((size_t)bm * 16) * C + c;
  float mx = -3.0e38f;
  #pragma unroll
  for (int kk = 0; kk < 16; ++kk) mx = fmaxf(mx, p[(size_t)kk * C]);
  out[i] = mx;
}
__global__ void pc_subsample(const float* __restrict__ src, float* __restrict__ dst,
                             int np, int nsrc, int stride)
{
  int i = blockIdx.x * 256 + threadIdx.x;
  if (i >= 2 * np * 3) return;
  int b = i / (np * 3), j = i % (np * 3), m2 = j / 3, c = j % 3;
  dst[i] = src[((size_t)b * nsrc + (size_t)m2 * stride) * 3 + c];
}
__global__ void pc_build_g(const float* __restrict__ rxyz, const float* __restrict__ qxyz,
                           const float* __restrict__ fsrc, const int* __restrict__ idx,
                           float* __restrict__ g, long R, int cin, int Cf, int np, int nsrc)
{
  long i = (long)blockIdx.x * 256 + threadIdx.x;
  if (i >= R * (long)cin) return;
  long r = i / cin; int c = (int)(i % cin);
  long bm = r >> 4;
  int b = (int)(bm / np), m2 = (int)(bm % np);
  int id = idx[r];
  if (c < 3)
    g[i] = rxyz[((size_t)b * nsrc + id) * 3 + c] - qxyz[((size_t)b * np + m2) * 3 + c];
  else
    g[i] = fsrc[((size_t)b * nsrc + id) * Cf + (c - 3)];
}
__global__ void pc_ybuild(float* __restrict__ y, const float* __restrict__ noise, int n) {
  int i = blockIdx.x * 256 + threadIdx.x;
  if (i < n) y[i] = y[i] + noise[i] - 0.5f;
}

// ------------------------- entropy bottleneck ------------------------------
struct EBW { const float* m[5]; const float* b[5]; const float* f[4]; };

__device__ __forceinline__ float sp_(float x)   { return fmaxf(x, 0.f) + log1pf(__expf(-fabsf(x))); }
__device__ __forceinline__ float sigm_(float x) { return 1.f / (1.f + __expf(-x)); }

__device__ float eb_chain(float x, int c, const EBW& w)
{
  float h0[3];
  #pragma unroll
  for (int j = 0; j < 3; ++j) {
    float v = sp_(w.m[0][c*3+j]) * x + w.b[0][c*3+j];
    v += tanhf(w.f[0][c*3+j]) * tanhf(v);
    h0[j] = v;
  }
  #pragma unroll
  for (int L = 1; L < 4; ++L) {
    float t[3];
    #pragma unroll
    for (int j = 0; j < 3; ++j) {
      float v = w.b[L][c*3+j];
      #pragma unroll
      for (int kk = 0; kk < 3; ++kk) v += sp_(w.m[L][(c*3+j)*3+kk]) * h0[kk];
      v += tanhf(w.f[L][c*3+j]) * tanhf(v);
      t[j] = v;
    }
    #pragma unroll
    for (int j = 0; j < 3; ++j) h0[j] = t[j];
  }
  float out = w.b[4][c];
  #pragma unroll
  for (int kk = 0; kk < 3; ++kk) out += sp_(w.m[4][c*3+kk]) * h0[kk];
  return out;
}

__global__ void pc_eb_bpp(const float* __restrict__ y, EBW w, float* __restrict__ acc,
                          int total, int nsamp)
{
  __shared__ float red[256];
  int i = blockIdx.x * 256 + threadIdx.x;
  float contrib = 0.f;
  if (i < total) {
    int c  = i / nsamp;
    int nn = i % nsamp;
    float x  = y[(size_t)nn * 16 + c];
    float lo = eb_chain(x - 0.5f, c, w);
    float up = eb_chain(x + 0.5f, c, w);
    float z  = lo + up;
    float sg = (z > 0.f) ? -1.f : ((z < 0.f) ? 1.f : 0.f);
    float lik = fabsf(sigm_(sg * up) - sigm_(sg * lo));
    lik = fmaxf(lik, 1e-9f);
    contrib = log2f(lik);
  }
  red[threadIdx.x] = contrib;
  __syncthreads();
  for (int s2 = 128; s2 > 0; s2 >>= 1) {
    if (threadIdx.x < s2) red[threadIdx.x] += red[threadIdx.x + s2];
    __syncthreads();
  }
  if (threadIdx.x == 0) atomicAdd(acc, red[0]);
}

__global__ void pc_eb_aux(const float* __restrict__ quant, EBW w, float* __restrict__ acc)
{
  int i = threadIdx.x;
  if (i >= 48) return;
  int c = i / 3, nn = i % 3;
  float x   = quant[c * 3 + nn];
  float out = eb_chain(x, c, w);
  const float t = 21.4164265f; // log(2/1e-9 - 1)
  float target = (nn == 0) ? -t : ((nn == 1) ? 0.f : t);
  atomicAdd(acc, fabsf(out - target));
}

__global__ void pc_finalize(const float* __restrict__ acc, float* __restrict__ out)
{
  if (threadIdx.x == 0) {
    float rec = (acc[0] + acc[1]) / (2.f * 8192.f);
    float bpp = acc[2] * (-1.0f / 16384.f);     // -BPP_F / (B*N)
    out[0] = rec + bpp + acc[3];                // AUX_F = 1
  }
}

// ------------------------------ host driver --------------------------------
struct TP { const float *fc1W,*fc1b,*fc2W,*fc2b,*wq,*wk,*wv,
                         *d1W,*d1b,*d2W,*d2b,*g1W,*g1b,*g2W,*g2b; };

static inline unsigned gb(long n) { return (unsigned)((n + 255) / 256); }

extern "C" void kernel_launch(void* const* d_in, const int* in_sizes, int n_in,
                              void* d_out, int out_size, void* d_ws, size_t ws_size,
                              hipStream_t stream)
{
  (void)in_sizes; (void)n_in; (void)out_size; (void)ws_size;
  auto F = [&](int i) -> const float* { return (const float*)d_in[i]; };

  // Flat leaf order of setup_inputs() (dict insertion order):
  // 0:x  1,2:mlp0a  3,4:mlp0b  5..19:t0(fc1,fc2,wq,wk,wv,d1,d2,g1,g2)
  // 20..27:tds  28..42:ts0  43..57:ts1  58,59:mlp1
  // 60..64:eb.mats  65..69:eb.biases  70..73:eb.factors  74:eb.quantiles
  // 75,76:dec.lin0  77..80:dec.tus  81,82:dec.lin1  83:noise
  const float* x     = F(0);
  const float* noise = F(83);
  auto mkTP = [&](int base) {
    TP t;
    t.fc1W=F(base+0);  t.fc1b=F(base+1);  t.fc2W=F(base+2);  t.fc2b=F(base+3);
    t.wq  =F(base+4);  t.wk  =F(base+5);  t.wv  =F(base+6);
    t.d1W =F(base+7);  t.d1b =F(base+8);  t.d2W =F(base+9);  t.d2b =F(base+10);
    t.g1W =F(base+11); t.g1b =F(base+12); t.g2W =F(base+13); t.g2b =F(base+14);
    return t;
  };
  const TP t0  = mkTP(5);
  const TP ts0 = mkTP(28);
  const TP ts1 = mkTP(43);
  EBW ebw;
  for (int i = 0; i < 5; ++i) { ebw.m[i] = F(60 + i); ebw.b[i] = F(65 + i); }
  for (int i = 0; i < 4; ++i)   ebw.f[i] = F(70 + i);

  // workspace bump allocator
  char* wp = (char*)d_ws;
  auto alloc = [&](size_t nfloats) -> float* {
    float* p = (float*)wp;
    wp += ((nfloats * sizeof(float) + 255) & ~(size_t)255);
    return p;
  };
  float* acc  = alloc(8);
  float* fA   = alloc(1u<<20);
  float* fB   = alloc(1u<<20);
  float* xbuf = alloc(1u<<20);
  float* qb   = alloc(1u<<20);
  float* kbuf = alloc(1u<<20);
  float* vbuf = alloc(1u<<20);
  float* obuf = alloc(1u<<20);
  int*   idxb = (int*)alloc(262144);
  float* relb = alloc(786432);
  float* kg   = alloc(16777216);
  float* vg   = alloc(16777216);
  float* posb = alloc(16777216);
  float* tmpb = alloc(16777216);
  float* xyz1 = alloc(12288);
  float* xyz2 = alloc(3072);
  float* ency = alloc(16384);

  auto gemm = [&](const float* A, const float* W, const float* bi, const float* Res,
                  float* C, long M, int K, int N, int act) {
    dim3 g((unsigned)((M + 63) / 64), (unsigned)((N + 127) / 128));
    if (act) pc_gemm_wmma<1><<<g, 256, 0, stream>>>(A, W, bi, Res, C, (int)M, K, N);
    else     pc_gemm_wmma<0><<<g, 256, 0, stream>>>(A, W, bi, Res, C, (int)M, K, N);
  };

  auto transformer = [&](const float* xyz, const float* fin, float* fout,
                         int Mpts, int Cf, int dm, const TP& t) {
    const int  BM = 2 * Mpts;
    const long R  = (long)BM * 16;
    pc_knn_topk<<<dim3(gb(Mpts), 2), 256, 0, stream>>>(xyz, xyz, Mpts, Mpts, idxb);
    gemm(fin,  t.fc1W, t.fc1b, nullptr, xbuf, BM, Cf, dm, 0);
    gemm(xbuf, t.wq,   nullptr, nullptr, qb,   BM, dm, dm, 0);
    gemm(xbuf, t.wk,   nullptr, nullptr, kbuf, BM, dm, dm, 0);
    gemm(xbuf, t.wv,   nullptr, nullptr, vbuf, BM, dm, dm, 0);
    pc_gather<<<gb(R * dm), 256, 0, stream>>>(kbuf, idxb, kg, R, dm, Mpts);
    pc_gather<<<gb(R * dm), 256, 0, stream>>>(vbuf, idxb, vg, R, dm, Mpts);
    pc_rel<<<gb(R * 3), 256, 0, stream>>>(xyz, idxb, relb, R, Mpts);
    gemm(relb, t.d1W, t.d1b, nullptr, tmpb, R, 3,  dm, 1);
    gemm(tmpb, t.d2W, t.d2b, nullptr, posb, R, dm, dm, 0);
    pc_add<<<gb(R * dm), 256, 0, stream>>>(vg, posb, R * (long)dm);       // vp = v + pos
    pc_qkp<<<gb(R * dm), 256, 0, stream>>>(kg, qb, posb, R, dm);          // q - k + pos
    gemm(kg,   t.g1W, t.g1b, nullptr, tmpb, R, dm, dm, 1);
    gemm(tmpb, t.g2W, t.g2b, nullptr, posb, R, dm, dm, 0);                // a
    pc_attend<<<gb((long)BM * dm), 256, 0, stream>>>(posb, vg, obuf, BM, dm);
    gemm(obuf, t.fc2W, t.fc2b, fin, fout, BM, dm, Cf, 0);                 // +residual
  };

  auto set_abstraction = [&](const float* xyz_src, const float* f_src, int Msrc, int Cf,
                             float* xyz_dst, float* f_dst, int np, int Cout,
                             const float* m1W, const float* m1b,
                             const float* m2W, const float* m2b) {
    const int  stride = Msrc / np;
    const long R = (long)2 * np * 16;
    const int  cin = 3 + Cf;
    pc_subsample<<<gb((long)2 * np * 3), 256, 0, stream>>>(xyz_src, xyz_dst, np, Msrc, stride);
    pc_knn_topk<<<dim3(gb(np), 2), 256, 0, stream>>>(xyz_dst, xyz_src, np, Msrc, idxb);
    pc_build_g<<<gb(R * cin), 256, 0, stream>>>(xyz_src, xyz_dst, f_src, idxb,
                                                kg, R, cin, Cf, np, Msrc);
    gemm(kg, m1W, m1b, nullptr, vg,   R, cin,  Cout, 1);
    gemm(vg, m2W, m2b, nullptr, posb, R, Cout, Cout, 1);
    pc_maxk<<<gb((long)2 * np * Cout), 256, 0, stream>>>(posb, f_dst, (long)2 * np, Cout);
  };

  // ---- forward pass ----
  pc_zero<<<1, 32, 0, stream>>>(acc, 8);

  // mlp0: relu(x@mlp0a)@mlp0b
  gemm(x,    F(1), F(2), nullptr, xbuf, 16384, 3,  64, 1);
  gemm(xbuf, F(3), F(4), nullptr, fA,   16384, 64, 64, 0);

  transformer(x, fA, fB, 8192, 64, 64, t0);                               // f -> fB
  set_abstraction(x, fB, 8192, 64, xyz1, fA, 2048, 128, F(20), F(21), F(22), F(23));
  transformer(xyz1, fA, fB, 2048, 128, 256, ts0);                         // f -> fB
  set_abstraction(xyz1, fB, 2048, 128, xyz2, fA, 512, 256, F(24), F(25), F(26), F(27));
  transformer(xyz2, fA, fB, 512, 256, 256, ts1);                          // f -> fB

  // encoder head: enc = f @ mlp1 + b;  y = enc + noise - 0.5
  gemm(fB, F(58), F(59), nullptr, ency, 1024, 256, 16, 0);
  pc_ybuild<<<gb(16384), 256, 0, stream>>>(ency, noise, 16384);

  // entropy bottleneck: bpp + aux
  pc_eb_bpp<<<gb(16384), 256, 0, stream>>>(ency, ebw, acc + 2, 16384, 1024);
  pc_eb_aux<<<1, 64, 0, stream>>>(F(74), ebw, acc + 3);

  // decoder: lin0 -> tus0 (reshape) -> tus1 (reshape) -> lin1
  gemm(ency, F(75), F(76), nullptr, qb,   1024,  16,  256, 0);
  gemm(qb,   F(77), F(78), nullptr, kbuf, 1024,  256, 512, 1);
  gemm(kbuf, F(79), F(80), nullptr, vbuf, 4096,  128, 256, 1);
  gemm(vbuf, F(81), F(82), nullptr, obuf, 16384, 64,  3,   0);

  // Chamfer(dec_out, x) both directions
  pc_chamfer_min<<<dim3(32, 2), 256, 0, stream>>>(obuf, x, 8192, 8192, acc + 0);
  pc_chamfer_min<<<dim3(32, 2), 256, 0, stream>>>(x, obuf, 8192, 8192, acc + 1);

  pc_finalize<<<1, 32, 0, stream>>>(acc, (float*)d_out);
}
